// EncoderBlock_31464930410694
// MI455X (gfx1250) — matrix operand order repaired and checked by reading, not compiled, not run on hardware
//
#include <hip/hip_runtime.h>
#include <cstdint>

typedef __attribute__((ext_vector_type(16))) __bf16 v16bf;
typedef __attribute__((ext_vector_type(8)))  float  v8f;
typedef unsigned short u16;
typedef unsigned int   u32;

union Frag {              // 16 bf16 = 32 bytes = 8 VGPRs
    uint4 q[2];
    v16bf v;
};

static __device__ __forceinline__ u16 f2bf(float f) {
    u32 u = __float_as_uint(f);
    u32 r = (u + 0x7FFFu + ((u >> 16) & 1u)) >> 16;   // round-to-nearest-even
    return (u16)r;
}

static __device__ __forceinline__ v8f wmma_bf16(v16bf a, v16bf b, v8f c) {
    return __builtin_amdgcn_wmma_f32_16x16x32_bf16(false, a, false, b, (short)0, c, false, false);
}

static __device__ __forceinline__ float rmax16(float x) {
    #pragma unroll
    for (int m = 1; m < 16; m <<= 1) x = fmaxf(x, __shfl_xor(x, m));
    return x;
}
static __device__ __forceinline__ float rsum16(float x) {
    #pragma unroll
    for (int m = 1; m < 16; m <<= 1) x += __shfl_xor(x, m);
    return x;
}

// CDNA5 async global->LDS copy (ASYNCcnt-tracked), 16B per lane.
static __device__ __forceinline__ void async_copy_b128(u32 lds_off, const void* gaddr) {
    asm volatile("global_load_async_to_lds_b128 %0, %1, off"
                 :: "v"(lds_off), "v"(gaddr)
                 : "memory");
}
static __device__ __forceinline__ void wait_async0() {
    asm volatile("s_wait_asynccnt 0x0" ::: "memory");
}

// ---------------------------------------------------------------- cast f32->bf16
__global__ __launch_bounds__(256) void cast_bf16_kernel(const float* __restrict__ in,
                                                        u16* __restrict__ out, int n) {
    int i = blockIdx.x * 256 + threadIdx.x;
    if (i < n) out[i] = f2bf(in[i]);
}

// ---------------------------------------------------------------- cast + transpose
// in: f32 [K,N] row-major. out: bf16 [N,K] (N-major). K%32==0, N%32==0.
__global__ __launch_bounds__(256) void cast_transpose_kernel(const float* __restrict__ in,
                                                             u16* __restrict__ out,
                                                             int K, int N) {
    __shared__ u16 tile[32][33];
    const int k0 = blockIdx.x * 32, n0 = blockIdx.y * 32;
    const int tx = threadIdx.x & 31, ty = threadIdx.x >> 5;   // 32 x 8
    #pragma unroll
    for (int i = 0; i < 32; i += 8)
        tile[ty + i][tx] = f2bf(in[(size_t)(k0 + ty + i) * N + n0 + tx]);
    __syncthreads();
    #pragma unroll
    for (int i = 0; i < 32; i += 8)
        out[(size_t)(n0 + ty + i) * K + k0 + tx] = tile[tx][ty + i];
}

// ---------------------------------------------------------------- WMMA GEMM
// C[M,N] = A[M,K] @ B[K,N] (+bias)(+relu); A bf16 row-major, Bt = B^T bf16 [N,K].
// M%128==0, N%128==0, K%32==0. Async double-buffered LDS tiles.
template<bool BIAS, bool RELU, bool OUTBF16>
__global__ __launch_bounds__(256) void gemm_wmma(const u16* __restrict__ A,
                                                 const u16* __restrict__ Bt,
                                                 const float* __restrict__ bias,
                                                 void* __restrict__ Cout,
                                                 int M, int N, int K) {
    constexpr int BM = 128, BN = 128, BK = 32;
    constexpr int LDA = BK + 8;                // 40 halves = 80B, bank-conflict-free
    __shared__ u16 As[2][BM * LDA];            // M-major
    __shared__ u16 Bs[2][BN * LDA];            // N-major

    const int tid  = threadIdx.x;
    const int lane = tid & 31, wid = tid >> 5;
    const int g = lane >> 4, l16 = lane & 15;
    const int bm = blockIdx.y, bn = blockIdx.x;
    const int waveM = wid & 3, waveN = wid >> 2;   // 4 x 2 wave grid
    const int n0 = bn * BN + waveN * 64;

    const u32 asBase[2] = { (u32)(uintptr_t)&As[0][0], (u32)(uintptr_t)&As[1][0] };
    const u32 bsBase[2] = { (u32)(uintptr_t)&Bs[0][0], (u32)(uintptr_t)&Bs[1][0] };

    // this thread's two 8-element chunks of the 128x32 tile
    const int c0r = tid >> 2,          c0c = (tid & 3) * 8;
    const int c1r = (tid + 256) >> 2,  c1c = ((tid + 256) & 3) * 8;
    const u16* aRow0 = A  + (size_t)(bm * BM + c0r) * K + c0c;
    const u16* aRow1 = A  + (size_t)(bm * BM + c1r) * K + c1c;
    const u16* bRow0 = Bt + (size_t)(bn * BN + c0r) * K + c0c;
    const u16* bRow1 = Bt + (size_t)(bn * BN + c1r) * K + c1c;
    const u32 lA0 = (c0r * LDA + c0c) * 2, lA1 = (c1r * LDA + c1c) * 2;

    auto issueTile = [&](int k0, int buf) {
        async_copy_b128(asBase[buf] + lA0, aRow0 + k0);
        async_copy_b128(asBase[buf] + lA1, aRow1 + k0);
        async_copy_b128(bsBase[buf] + lA0, bRow0 + k0);
        async_copy_b128(bsBase[buf] + lA1, bRow1 + k0);
    };

    v8f acc[2][4];
    const v8f zf = {0.f,0.f,0.f,0.f,0.f,0.f,0.f,0.f};
    #pragma unroll
    for (int i = 0; i < 2; ++i)
        #pragma unroll
        for (int j = 0; j < 4; ++j) acc[i][j] = zf;

    issueTile(0, 0);
    wait_async0();
    __syncthreads();

    int buf = 0;
    for (int k0 = 0; k0 < K; k0 += BK) {
        if (k0 + BK < K) issueTile(k0 + BK, buf ^ 1);   // prefetch next tile

        Frag af[2], bf[4];
        #pragma unroll
        for (int mt = 0; mt < 2; ++mt) {
            int r = waveM * 32 + mt * 16 + l16;
            af[mt].q[0] = *(const uint4*)(&As[buf][r * LDA + g * 8]);
            af[mt].q[1] = *(const uint4*)(&As[buf][r * LDA + 16 + g * 8]);
        }
        #pragma unroll
        for (int nt = 0; nt < 4; ++nt) {
            int n = waveN * 64 + nt * 16 + l16;
            bf[nt].q[0] = *(const uint4*)(&Bs[buf][n * LDA + g * 8]);
            bf[nt].q[1] = *(const uint4*)(&Bs[buf][n * LDA + 16 + g * 8]);
        }
        #pragma unroll
        for (int mt = 0; mt < 2; ++mt)
            #pragma unroll
            for (int nt = 0; nt < 4; ++nt)
                acc[mt][nt] = wmma_bf16(af[mt].v, bf[nt].v, acc[mt][nt]);

        wait_async0();          // next tile resident (this wave)
        __syncthreads();        // all waves done reading `buf`, next tile published
        buf ^= 1;
    }

    // epilogue: C layout is M = v + 8*g, N = l16 within each 16x16 tile
    const int m0 = bm * BM + waveM * 32;
    #pragma unroll
    for (int mt = 0; mt < 2; ++mt)
        #pragma unroll
        for (int nt = 0; nt < 4; ++nt) {
            int col = n0 + nt * 16 + l16;
            float bv = BIAS ? bias[col] : 0.f;
            #pragma unroll
            for (int v = 0; v < 8; ++v) {
                int row = m0 + mt * 16 + v + 8 * g;
                float val = acc[mt][nt][v] + bv;
                if (RELU) val = fmaxf(val, 0.f);
                if (OUTBF16) ((u16*)Cout)[(size_t)row * N + col] = f2bf(val);
                else         ((float*)Cout)[(size_t)row * N + col] = val;
            }
        }
}

// ---------------------------------------------------------------- flash attention
// Q,K,V: bf16 [B*S, 1024] (head h at cols h*64..h*64+63). Out: bf16 same shape.
__global__ __launch_bounds__(256) void attn_kernel(const u16* __restrict__ Q,
                                                   const u16* __restrict__ Km,
                                                   const u16* __restrict__ Vm,
                                                   const int* __restrict__ mask,
                                                   u16* __restrict__ Out) {
    constexpr int S = 2048, DM = 1024, DK = 64, NH = 16, TKV = 64;
    constexpr int LDK = DK + 8;                 // 72 halves = 144B
    __shared__ u16 Kt[TKV * LDK];               // [key][d]      (N-major for Q@K^T)
    __shared__ u16 Vt[DK * LDK];                // [d][key]      (N-major for P@V)
    __shared__ u16 Pw[8][16 * LDK];             // per-wave P tile 16x64 (+pad)

    const int tid  = threadIdx.x;
    const int lane = tid & 31, wid = tid >> 5;
    const int g = lane >> 4, l16 = lane & 15;
    const int bh = blockIdx.y;
    const int b = bh / NH, h = bh % NH;
    const int q0 = blockIdx.x * 128 + wid * 16;    // wave's 16 queries
    const size_t rowBase = (size_t)b * S;
    const u32 ktBase = (u32)(uintptr_t)&Kt[0];

    // Q A-fragments for the whole d=64 (two 16x32 k-chunks), kept in registers
    Frag qf[2];
    const u16* qrow = Q + (rowBase + q0 + l16) * DM + h * DK;
    #pragma unroll
    for (int kc = 0; kc < 2; ++kc) {
        qf[kc].q[0] = *(const uint4*)(qrow + kc * 32 + g * 8);
        qf[kc].q[1] = *(const uint4*)(qrow + kc * 32 + 16 + g * 8);
    }

    const v8f zf = {0.f,0.f,0.f,0.f,0.f,0.f,0.f,0.f};
    float mi[8], li[8];
    v8f o[4];
    #pragma unroll
    for (int v = 0; v < 8; ++v) { mi[v] = -1e30f; li[v] = 0.f; }
    #pragma unroll
    for (int t = 0; t < 4; ++t) o[t] = zf;

    for (int kv0 = 0; kv0 < S; kv0 += TKV) {
        __syncthreads();                        // prior readers of Kt/Vt done
        // K tile via async copy; V tile transposed through VGPRs
        #pragma unroll
        for (int i = 0; i < 2; ++i) {
            int c = tid + i * 256;              // 512 chunks of 8
            int r = c >> 3, c8 = (c & 7) * 8;
            async_copy_b128(ktBase + (r * LDK + c8) * 2,
                            Km + (rowBase + kv0 + r) * DM + h * DK + c8);
            uint4 dv = *(const uint4*)(Vm + (rowBase + kv0 + r) * DM + h * DK + c8);
            const u16* ds = (const u16*)&dv;
            #pragma unroll
            for (int j = 0; j < 8; ++j) Vt[(c8 + j) * LDK + r] = ds[j];
        }
        wait_async0();
        __syncthreads();

        // S = Q @ K^T  (per wave: 16 queries x 64 keys)
        v8f s[4];
        #pragma unroll
        for (int t = 0; t < 4; ++t) s[t] = zf;
        #pragma unroll
        for (int kc = 0; kc < 2; ++kc) {
            #pragma unroll
            for (int t = 0; t < 4; ++t) {
                Frag kb;
                int n = t * 16 + l16;
                kb.q[0] = *(const uint4*)(&Kt[n * LDK + kc * 32 + g * 8]);
                kb.q[1] = *(const uint4*)(&Kt[n * LDK + kc * 32 + 16 + g * 8]);
                s[t] = wmma_bf16(qf[kc].v, kb.v, s[t]);
            }
        }

        // scale + mask
        #pragma unroll
        for (int t = 0; t < 4; ++t) {
            int kg = kv0 + t * 16 + l16;
            #pragma unroll
            for (int v = 0; v < 8; ++v) {
                int qg = q0 + v + 8 * g;
                float x = s[t][v] * 0.125f;     // 1/sqrt(64)
                if (mask[((size_t)b * S + qg) * S + kg] == 0) x = -1.0e9f;
                s[t][v] = x;
            }
        }

        // online softmax per row (rows live in 16-lane halves)
        #pragma unroll
        for (int v = 0; v < 8; ++v) {
            float r = fmaxf(fmaxf(s[0][v], s[1][v]), fmaxf(s[2][v], s[3][v]));
            r = rmax16(r);
            float mnew = fmaxf(mi[v], r);
            float alpha = __expf(mi[v] - mnew);
            mi[v] = mnew;
            float rs = 0.f;
            #pragma unroll
            for (int t = 0; t < 4; ++t) {
                float p = __expf(s[t][v] - mnew);
                s[t][v] = p;
                rs += p;
            }
            rs = rsum16(rs);
            li[v] = li[v] * alpha + rs;
            #pragma unroll
            for (int t = 0; t < 4; ++t) o[t][v] *= alpha;
        }

        // C-layout -> A-fragment layout via per-wave LDS (in-order within wave)
        #pragma unroll
        for (int t = 0; t < 4; ++t)
            #pragma unroll
            for (int v = 0; v < 8; ++v)
                Pw[wid][(v + 8 * g) * LDK + t * 16 + l16] = f2bf(s[t][v]);
        asm volatile("" ::: "memory");

        // O += P @ V
        #pragma unroll
        for (int kc = 0; kc < 2; ++kc) {
            Frag pf;
            pf.q[0] = *(const uint4*)(&Pw[wid][l16 * LDK + kc * 32 + g * 8]);
            pf.q[1] = *(const uint4*)(&Pw[wid][l16 * LDK + kc * 32 + 16 + g * 8]);
            #pragma unroll
            for (int t2 = 0; t2 < 4; ++t2) {
                Frag vb;
                int n = t2 * 16 + l16;
                vb.q[0] = *(const uint4*)(&Vt[n * LDK + kc * 32 + g * 8]);
                vb.q[1] = *(const uint4*)(&Vt[n * LDK + kc * 32 + 16 + g * 8]);
                o[t2] = wmma_bf16(pf.v, vb.v, o[t2]);
            }
        }
    }

    #pragma unroll
    for (int t = 0; t < 4; ++t)
        #pragma unroll
        for (int v = 0; v < 8; ++v) {
            int qg = q0 + v + 8 * g;
            float val = o[t][v] / li[v];
            Out[(rowBase + qg) * DM + h * DK + t * 16 + l16] = f2bf(val);
        }
}

// ---------------------------------------------------------------- residual + LayerNorm
// y = alpha*(x - mean)/(std(ddof=1) + eps) + beta, per 1024-elem row, x = X + Y
template<bool EMIT_BF16>
__global__ __launch_bounds__(256) void ln_res_kernel(const float* __restrict__ X,
                                                     const float* __restrict__ Y,
                                                     const float* __restrict__ alpha,
                                                     const float* __restrict__ beta,
                                                     float* __restrict__ OutF,
                                                     u16* __restrict__ OutB) {
    constexpr int D = 1024;
    __shared__ float red[256];
    const int tid = threadIdx.x;
    const size_t base = (size_t)blockIdx.x * D;

    float v[4];
    float s = 0.f;
    #pragma unroll
    for (int i = 0; i < 4; ++i) {
        int c = tid + i * 256;
        v[i] = X[base + c] + Y[base + c];
        s += v[i];
    }
    red[tid] = s; __syncthreads();
    for (int o = 128; o > 0; o >>= 1) { if (tid < o) red[tid] += red[tid + o]; __syncthreads(); }
    float mean = red[0] * (1.f / D);
    __syncthreads();

    float sq = 0.f;
    #pragma unroll
    for (int i = 0; i < 4; ++i) { float d = v[i] - mean; sq += d * d; }
    red[tid] = sq; __syncthreads();
    for (int o = 128; o > 0; o >>= 1) { if (tid < o) red[tid] += red[tid + o]; __syncthreads(); }
    float var = red[0] * (1.f / (D - 1));       // Bessel-corrected (torch default)
    float inv = alpha[0] / (sqrtf(var) + 1e-6f);
    float bt  = beta[0];

    #pragma unroll
    for (int i = 0; i < 4; ++i) {
        int c = tid + i * 256;
        float r = (v[i] - mean) * inv + bt;
        OutF[base + c] = r;
        if (EMIT_BF16) OutB[base + c] = f2bf(r);
    }
}

// ---------------------------------------------------------------- host launcher
extern "C" void kernel_launch(void* const* d_in, const int* in_sizes, int n_in,
                              void* d_out, int out_size, void* d_ws, size_t ws_size,
                              hipStream_t stream) {
    (void)in_sizes; (void)n_in; (void)out_size; (void)ws_size;
    constexpr int Bq = 4, S = 2048, DM = 1024, DFF = 4096;
    constexpr int Mrows = Bq * S;               // 8192

    const float* x    = (const float*)d_in[0];
    const int*   mask = (const int*)  d_in[1];
    const float* wq   = (const float*)d_in[2];
    const float* wk   = (const float*)d_in[3];
    const float* wv   = (const float*)d_in[4];
    const float* wo   = (const float*)d_in[5];
    const float* w1   = (const float*)d_in[6];
    const float* b1   = (const float*)d_in[7];
    const float* w2   = (const float*)d_in[8];
    const float* b2   = (const float*)d_in[9];
    const float* a1   = (const float*)d_in[10];
    const float* bt1  = (const float*)d_in[11];
    const float* a2   = (const float*)d_in[12];
    const float* bt2  = (const float*)d_in[13];

    char* w = (char*)d_ws;
    constexpr size_t MB = 1024ull * 1024ull;
    u16*  XB    = (u16*)(w + 0);          // 16 MB
    u16*  WQT   = (u16*)(w + 16 * MB);    //  2 MB each, transposed [N,K] bf16
    u16*  WKT   = (u16*)(w + 18 * MB);
    u16*  WVT   = (u16*)(w + 20 * MB);
    u16*  WOT   = (u16*)(w + 22 * MB);
    u16*  W1T   = (u16*)(w + 24 * MB);    //  8 MB  [4096,1024]
    u16*  W2T   = (u16*)(w + 32 * MB);    //  8 MB  [1024,4096]
    u16*  QB    = (u16*)(w + 40 * MB);    // 16 MB
    u16*  KB    = (u16*)(w + 56 * MB);    // 16 MB
    u16*  VB    = (u16*)(w + 72 * MB);    // 16 MB
    u16*  ATTNB = (u16*)(w + 88 * MB);    // 16 MB
    u16*  F1B   = (u16*)(w + 40 * MB);    // 64 MB, aliases Q/K/V/ATTN (dead by then)
    float* PROJ = (float*)(w + 104 * MB); // 32 MB
    float* F2F  = (float*)(w + 104 * MB); // 32 MB, aliases PROJ (dead by then)
    float* H1F  = (float*)(w + 136 * MB); // 32 MB
    u16*  H1B   = (u16*)(w + 168 * MB);   // 16 MB  -> total 184 MB

    // 1) x -> bf16; weights -> transposed bf16 (N-major)
    {
        int n = Mrows * DM;
        cast_bf16_kernel<<<n / 256, 256, 0, stream>>>(x, XB, n);
        dim3 gT(DM / 32, DM / 32);
        cast_transpose_kernel<<<gT, 256, 0, stream>>>(wq, WQT, DM, DM);
        cast_transpose_kernel<<<gT, 256, 0, stream>>>(wk, WKT, DM, DM);
        cast_transpose_kernel<<<gT, 256, 0, stream>>>(wv, WVT, DM, DM);
        cast_transpose_kernel<<<gT, 256, 0, stream>>>(wo, WOT, DM, DM);
        cast_transpose_kernel<<<dim3(DM / 32, DFF / 32), 256, 0, stream>>>(w1, W1T, DM, DFF);
        cast_transpose_kernel<<<dim3(DFF / 32, DM / 32), 256, 0, stream>>>(w2, W2T, DFF, DM);
    }

    // 2) Q, K, V projections
    dim3 gProj(DM / 128, Mrows / 128);
    gemm_wmma<false, false, true><<<gProj, 256, 0, stream>>>(XB, WQT, nullptr, QB, Mrows, DM, DM);
    gemm_wmma<false, false, true><<<gProj, 256, 0, stream>>>(XB, WKT, nullptr, KB, Mrows, DM, DM);
    gemm_wmma<false, false, true><<<gProj, 256, 0, stream>>>(XB, WVT, nullptr, VB, Mrows, DM, DM);

    // 3) flash attention
    attn_kernel<<<dim3(S / 128, Bq * 16), 256, 0, stream>>>(QB, KB, VB, mask, ATTNB);

    // 4) output projection (f32 out for residual)
    gemm_wmma<false, false, false><<<gProj, 256, 0, stream>>>(ATTNB, WOT, nullptr, PROJ, Mrows, DM, DM);

    // 5) residual + LN1 -> f32 + bf16
    ln_res_kernel<true><<<Mrows, 256, 0, stream>>>(x, PROJ, a1, bt1, H1F, H1B);

    // 6) FFN
    gemm_wmma<true, true, true><<<dim3(DFF / 128, Mrows / 128), 256, 0, stream>>>(
        H1B, W1T, b1, F1B, Mrows, DFF, DM);
    gemm_wmma<true, false, false><<<gProj, 256, 0, stream>>>(F1B, W2T, b2, F2F, Mrows, DM, DFF);

    // 7) residual + LN2 -> d_out (f32)
    ln_res_kernel<false><<<Mrows, 256, 0, stream>>>(H1F, F2F, a2, bt2, (float*)d_out, nullptr);
}
